// FunctionHead_83837761618384
// MI455X (gfx1250) — matrix operand order repaired and checked by reading, not compiled, not run on hardware
//
#include <hip/hip_runtime.h>

typedef __attribute__((ext_vector_type(16))) __bf16          v16bf;
typedef __attribute__((ext_vector_type(8)))  float           v8f;
typedef __attribute__((ext_vector_type(4)))  float           f32x4;
typedef __attribute__((ext_vector_type(8)))  unsigned short  u16x8;

// ---------------- helpers ----------------

__device__ __forceinline__ unsigned short f2bf(float f) {
    // round-to-nearest-even f32 -> bf16
    unsigned int u = __builtin_bit_cast(unsigned int, f);
    u += 0x7FFFu + ((u >> 16) & 1u);
    return (unsigned short)(u >> 16);
}

// WGP-scope prefetch: pulls line into all cache levels (ISA 10.5, scope 0).
// __builtin_prefetch can only reach SE/DEV/SYS scopes, which skip the WGP cache.
__device__ __forceinline__ void prefetch_wgp(const void* p) {
    asm volatile("global_prefetch_b8 %0, off offset:64" :: "v"(p));
}

union BfFrag {
    v16bf v;
    u16x8 h[2];
};

// A fragment (16x32 MxK bf16) from bf16 row-major source.
// p0 = &src[row*ld + k0 + (lane>=16 ? 8 : 0)]
// lanes<16: K=[k0..k0+7],[k0+16..k0+23]; lanes>=16: K=[k0+8..k0+15],[k0+24..k0+31]
__device__ __forceinline__ v16bf load_a_bf16(const unsigned short* __restrict__ p0) {
    BfFrag f;
    f.h[0] = *(const u16x8*)(p0);
    f.h[1] = *(const u16x8*)(p0 + 16);
    return f.v;
}

// B fragment (32x16 KxN bf16) from bf16 row-major W (N x K): column n = lane%16.
// p0 = &W[n*ld + k0 + (lane>=16 ? 8 : 0)]; elems = K [p0..p0+7],[p0+16..p0+23].
__device__ __forceinline__ v16bf load_b_bf16(const unsigned short* __restrict__ p0) {
    BfFrag f;
    f.h[0] = *(const u16x8*)(p0);
    f.h[1] = *(const u16x8*)(p0 + 16);
    return f.v;
}

// ---------------- f32 -> bf16 bulk conversion (memory bound) ----------------

__global__ __launch_bounds__(256) void cvt_bf16_kernel(
    const float* __restrict__ src, unsigned short* __restrict__ dst, long n)
{
    long i = ((long)blockIdx.x * 256 + threadIdx.x) * 8;
    if (i >= n) return;
    f32x4 x0 = *(const f32x4*)(src + i);
    f32x4 x1 = *(const f32x4*)(src + i + 4);
    u16x8 o;
#pragma unroll
    for (int c = 0; c < 4; ++c) { o[c] = f2bf(x0[c]); o[4 + c] = f2bf(x1[c]); }
    *(u16x8*)(dst + i) = o;
}

// ---------------- WMMA GEMM core ----------------
// Block: 256 threads = 8 waves as 2(M) x 4(N). Block tile 128 x 256.
// Wave tile 64 x 64 = 4x4 WMMA fragments. Pure-bf16 inner loop:
// 16 x global_load_b128 : 16 x v_wmma_f32_16x16x32_bf16 per K-step.

template <int K>
__device__ __forceinline__ void wmma_gemm_tile(
    const unsigned short* __restrict__ A,   // M x K bf16 row-major
    const unsigned short* __restrict__ B,   // N x K bf16 row-major
    v8f acc[4][4], int m0, int n0, int lane)
{
    const int lane16 = lane & 15;
    const int hi     = lane >> 4;
    const int aoff   = hi * 8;
    const int boff   = hi * 8;

    const unsigned short* pa[4];
    const unsigned short* pb[4];
#pragma unroll
    for (int i = 0; i < 4; ++i) pa[i] = A + (size_t)(m0 + i * 16 + lane16) * K + aoff;
#pragma unroll
    for (int j = 0; j < 4; ++j) pb[j] = B + (size_t)(n0 + j * 16 + lane16) * K + boff;

    for (int k0 = 0; k0 < K; k0 += 32) {
        // WGP-scope prefetch of the next K-tile into the near cache
        prefetch_wgp(pa[0] + k0);
        prefetch_wgp(pb[0] + k0);
        v16bf a[4], b[4];
#pragma unroll
        for (int i = 0; i < 4; ++i) a[i] = load_a_bf16(pa[i] + k0);
#pragma unroll
        for (int j = 0; j < 4; ++j) b[j] = load_b_bf16(pb[j] + k0);
#pragma unroll
        for (int i = 0; i < 4; ++i)
#pragma unroll
            for (int j = 0; j < 4; ++j)
                acc[i][j] = __builtin_amdgcn_wmma_f32_16x16x32_bf16(
                    false, a[i], false, b[j], (short)0, acc[i][j], false, false);
    }
}

// ---------- GEMM1: h = relu(inp_bf @ W1_bf^T + b1) -> bf16 ----------
// grid (2048/256, 16384/128) = (8, 128): N-tile fastest so concurrent blocks
// share the same inp rows in L2; W1 (16 MB bf16) stays L2-resident.

__global__ __launch_bounds__(256) void gemm1_relu_kernel(
    const unsigned short* __restrict__ inpb, const unsigned short* __restrict__ w1b,
    const float* __restrict__ b1, unsigned short* __restrict__ h)
{
    const int lane  = threadIdx.x & 31;
    const int wave  = threadIdx.x >> 5;
    const int m0 = blockIdx.y * 128 + (wave >> 2) * 64;
    const int n0 = blockIdx.x * 256 + (wave & 3) * 64;
    const int lane16 = lane & 15;
    const int hi     = lane >> 4;

    const v8f zero = {0.f, 0.f, 0.f, 0.f, 0.f, 0.f, 0.f, 0.f};
    v8f acc[4][4];
#pragma unroll
    for (int i = 0; i < 4; ++i)
#pragma unroll
        for (int j = 0; j < 4; ++j) acc[i][j] = zero;

    wmma_gemm_tile<4096>(inpb, w1b, acc, m0, n0, lane);

#pragma unroll
    for (int j = 0; j < 4; ++j) {
        const int col  = n0 + j * 16 + lane16;
        const float bias = b1[col];
#pragma unroll
        for (int i = 0; i < 4; ++i) {
#pragma unroll
            for (int r = 0; r < 8; ++r) {
                const int row = m0 + i * 16 + hi * 8 + r;
                float v = acc[i][j][r] + bias;
                v = v > 0.f ? v : 0.f;
                h[(size_t)row * 2048 + col] = f2bf(v);
            }
        }
    }
}

// ---------- GEMM2: logits = h @ W2_bf^T + b2 -> f32 ----------
// grid (512/256, 16384/128) = (2, 128)

__global__ __launch_bounds__(256) void gemm2_kernel(
    const unsigned short* __restrict__ h, const unsigned short* __restrict__ w2b,
    const float* __restrict__ b2, float* __restrict__ logits)
{
    const int lane  = threadIdx.x & 31;
    const int wave  = threadIdx.x >> 5;
    const int m0 = blockIdx.y * 128 + (wave >> 2) * 64;
    const int n0 = blockIdx.x * 256 + (wave & 3) * 64;
    const int lane16 = lane & 15;
    const int hi     = lane >> 4;

    const v8f zero = {0.f, 0.f, 0.f, 0.f, 0.f, 0.f, 0.f, 0.f};
    v8f acc[4][4];
#pragma unroll
    for (int i = 0; i < 4; ++i)
#pragma unroll
        for (int j = 0; j < 4; ++j) acc[i][j] = zero;

    wmma_gemm_tile<2048>(h, w2b, acc, m0, n0, lane);

#pragma unroll
    for (int j = 0; j < 4; ++j) {
        const int col  = n0 + j * 16 + lane16;
        const float bias = b2[col];
#pragma unroll
        for (int i = 0; i < 4; ++i) {
#pragma unroll
            for (int r = 0; r < 8; ++r) {
                const int row = m0 + i * 16 + hi * 8 + r;
                logits[(size_t)row * 512 + col] = acc[i][j][r] + bias;
            }
        }
    }
}

// ---------- top-8 gate + hard limit (logit space) ----------
// sigmoid(x) > 0.2  <=>  x > ln(0.2/0.8); top-8 via 8 rounds of bounded max.
// One wave per row; grid 2048, block 256 (8 rows per block).

__global__ __launch_bounds__(256) void topk_gate_kernel(
    const float* __restrict__ logits, float* __restrict__ out)
{
    const int lane = threadIdx.x & 31;
    const int wave = threadIdx.x >> 5;
    const int row  = blockIdx.x * 8 + wave;
    const float* p = logits + (size_t)row * 512;

    f32x4 x[4];
#pragma unroll
    for (int i = 0; i < 4; ++i)
        x[i] = *(const f32x4*)(p + i * 128 + lane * 4);

    float cur = __builtin_inff();
#pragma unroll
    for (int it = 0; it < 8; ++it) {
        float m = -__builtin_inff();
#pragma unroll
        for (int i = 0; i < 4; ++i)
#pragma unroll
            for (int c = 0; c < 4; ++c) {
                float v = x[i][c];
                if (v < cur) m = fmaxf(m, v);
            }
#pragma unroll
        for (int off = 16; off >= 1; off >>= 1)
            m = fmaxf(m, __shfl_xor(m, off, 32));
        cur = m;
    }
    const float kth = cur;
    const float LTH = -1.3862943611198906f;  // ln(0.2/0.8)

#pragma unroll
    for (int i = 0; i < 4; ++i) {
        f32x4 o;
#pragma unroll
        for (int c = 0; c < 4; ++c) {
            float v = x[i][c];
            o[c] = (v >= kth && v > LTH) ? 1.0f : 0.0f;
        }
        *(f32x4*)(out + (size_t)row * 512 + i * 128 + lane * 4) = o;
    }
}

// ---------------- launch ----------------

extern "C" void kernel_launch(void* const* d_in, const int* in_sizes, int n_in,
                              void* d_out, int out_size, void* d_ws, size_t ws_size,
                              hipStream_t stream) {
    (void)in_sizes; (void)n_in; (void)out_size; (void)ws_size;
    const float* inp = (const float*)d_in[0];   // 16384 x 4096
    const float* W1  = (const float*)d_in[1];   // 2048 x 4096
    const float* b1  = (const float*)d_in[2];   // 2048
    const float* W2  = (const float*)d_in[3];   // 512 x 2048
    const float* b2  = (const float*)d_in[4];   // 512
    // d_in[5] is k == 8 (hardcoded)

    const long N_INP = 16384L * 4096;   // 67,108,864
    const long N_W1  = 2048L * 4096;    //  8,388,608
    const long N_W2  = 512L * 2048;     //  1,048,576

    char* ws = (char*)d_ws;
    unsigned short* inpb   = (unsigned short*)(ws);                          // 128 MB
    unsigned short* w1b    = (unsigned short*)(ws + N_INP * 2);              //  16 MB
    unsigned short* w2b    = (unsigned short*)(ws + (N_INP + N_W1) * 2);     //   2 MB
    unsigned short* h      = (unsigned short*)(ws + (N_INP + N_W1 + N_W2) * 2);          // 64 MB
    float*          logits = (float*)(ws + (N_INP + N_W1 + N_W2 + 16384L * 2048) * 2);   // 32 MB
    float*          out    = (float*)d_out;

    dim3 blk(256);
    cvt_bf16_kernel<<<dim3((unsigned)(N_INP / 8 / 256)), blk, 0, stream>>>(inp, inpb, N_INP);
    cvt_bf16_kernel<<<dim3((unsigned)(N_W1 / 8 / 256)),  blk, 0, stream>>>(W1, w1b, N_W1);
    cvt_bf16_kernel<<<dim3((unsigned)(N_W2 / 8 / 256)),  blk, 0, stream>>>(W2, w2b, N_W2);

    gemm1_relu_kernel<<<dim3(8, 128), blk, 0, stream>>>(inpb, w1b, b1, h);
    gemm2_kernel<<<dim3(2, 128), blk, 0, stream>>>(h, w2b, b2, logits);
    topk_gate_kernel<<<dim3(2048), blk, 0, stream>>>(logits, out);
}
